// Attention_12652973654254
// MI455X (gfx1250) — matrix operand order repaired and checked by reading, not compiled, not run on hardware
//
#include <hip/hip_runtime.h>
#include <hip/hip_bf16.h>

typedef __attribute__((ext_vector_type(16))) _Float16 v16h;
typedef __attribute__((ext_vector_type(8)))  _Float16 v8h;
typedef __attribute__((ext_vector_type(8)))  float    v8f;

// Problem constants
constexpr int SEQ   = 128;
constexpr int BSZ   = 32;
constexpr int SB    = SEQ * BSZ;     // 4096 rows
constexpr int DIM   = 1024;
constexpr int NB    = 8;             // blocks
constexpr int BD    = 128;           // block dim
constexpr int NH    = 12;            // heads
constexpr int HD    = 64;            // head dim
constexpr int INNER = NH * HD;       // 768
constexpr float SCALE = 0.125f;      // 64^-0.5

// ---------------- WMMA fragment loaders ----------------
// A-matrix 16x32 f16 (row-major source, row stride lda in halves).
// Lane L: row = L&15, K = (L>>4)*8 + {0..7} and (L>>4)*8 + 16 + {0..7}.
__device__ inline v16h load_frag_a(const _Float16* __restrict__ base, int lda, int lane) {
    const int r  = lane & 15;
    const int hi = lane >> 4;
    const _Float16* p = base + (size_t)r * lda + hi * 8;
    union { v16h v; v8h h[2]; } u;
    u.h[0] = *(const v8h*)(p);        // K = hi*8 + 0..7
    u.h[1] = *(const v8h*)(p + 16);   // K = hi*8 + 16..23
    return u.v;
}

// B-matrix 32x16 f16 from transposed weights (N rows x K cols, row stride ldb).
// Lane L: col N = L&15, K = (L>>4)*8 + {0..7} and (L>>4)*8 + 16 + {0..7}.
__device__ inline v16h load_frag_b(const _Float16* __restrict__ base, int ldb, int lane) {
    const int n  = lane & 15;
    const int hi = lane >> 4;
    const _Float16* p = base + (size_t)n * ldb + hi * 8;
    union { v16h v; v8h h[2]; } u;
    u.h[0] = *(const v8h*)(p);
    u.h[1] = *(const v8h*)(p + 16);
    return u.v;
}

// ---------------- Prep kernels ----------------
__global__ void zero_f32(float* __restrict__ p, int n) {
    int i = blockIdx.x * blockDim.x + threadIdx.x;
    if (i < n) p[i] = 0.0f;
}

__global__ void cvt_f32_f16(const float* __restrict__ in, _Float16* __restrict__ out, int n) {
    int i = blockIdx.x * blockDim.x + threadIdx.x;
    if (i < n) out[i] = (_Float16)in[i];
}

// Wq/Wk/Wv: (8,128,768) -> transposed (8,768,128) f16
__global__ void transpose_w_qkv(const float* __restrict__ in, _Float16* __restrict__ out) {
    int i = blockIdx.x * blockDim.x + threadIdx.x;
    if (i >= NB * BD * INNER) return;
    int e = i % INNER;
    int t = i / INNER;
    int d = t % BD;
    int kb = t / BD;
    out[((size_t)kb * INNER + e) * BD + d] = (_Float16)in[i];
}

// Wf: (8,768,128) -> transposed (8,128,768) f16
__global__ void transpose_wf(const float* __restrict__ in, _Float16* __restrict__ out) {
    int i = blockIdx.x * blockDim.x + threadIdx.x;
    if (i >= NB * INNER * BD) return;
    int n = i % BD;
    int t = i / BD;
    int e = t % INNER;
    int kb = t / INNER;
    out[((size_t)kb * BD + n) * INNER + e] = (_Float16)in[i];
}

// ---------------- Fused QKV projection + attention ----------------
// grid = (SB/16, NH), block = 128 (4 waves). Wave w does blocks {2w, 2w+1}.
__global__ __launch_bounds__(128) void qkv_attn_kernel(
    const _Float16* __restrict__ xh,       // (4096, 1024)
    const _Float16* __restrict__ wqT,      // (8, 768, 128)
    const _Float16* __restrict__ wkT,
    const _Float16* __restrict__ wvT,
    _Float16* __restrict__ attnb,          // (4096, 8, 768)  (sb, block, h*64+d)
    float* __restrict__ smean)             // (4096, 8, 8)
{
    __shared__ _Float16 sq[16][NB][HD];
    __shared__ _Float16 sk[16][NB][HD];
    __shared__ _Float16 sv[16][NB][HD];

    const int rt   = blockIdx.x;           // row tile (16 sb rows)
    const int h    = blockIdx.y;           // head
    const int wave = threadIdx.x >> 5;
    const int lane = threadIdx.x & 31;
    const int r0   = rt * 16;

    // ---- GEMM phase: q,k,v head-slices via WMMA ----
    for (int bi = 0; bi < 2; ++bi) {
        const int kb = wave * 2 + bi;
        v16h afrag[4];
        #pragma unroll
        for (int ks = 0; ks < 4; ++ks)
            afrag[ks] = load_frag_a(xh + (size_t)r0 * DIM + kb * BD + ks * 32, DIM, lane);

        for (int m = 0; m < 3; ++m) {
            const _Float16* wT = (m == 0) ? wqT : (m == 1) ? wkT : wvT;
            const _Float16* wb = wT + ((size_t)kb * INNER + h * HD) * BD;
            const float scale  = (m == 0) ? SCALE : 1.0f;
            _Float16 (*dst)[NB][HD] = (m == 0) ? sq : (m == 1) ? sk : sv;

            #pragma unroll
            for (int nt = 0; nt < 4; ++nt) {
                v8f acc = {};
                #pragma unroll
                for (int ks = 0; ks < 4; ++ks) {
                    v16h bfrag = load_frag_b(wb + (size_t)(nt * 16) * BD + ks * 32, BD, lane);
                    acc = __builtin_amdgcn_wmma_f32_16x16x32_f16(
                        false, afrag[ks], false, bfrag, (short)0, acc, false, false);
                }
                const int col = lane & 15, hi = lane >> 4;
                #pragma unroll
                for (int g = 0; g < 8; ++g) {
                    const int row = hi * 8 + g;
                    dst[row][kb][nt * 16 + col] = (_Float16)(acc[g] * scale);
                }
            }
        }
    }
    __syncthreads();

    // ---- Attention phase: each thread owns one (row, q-block) ----
    const int r = threadIdx.x >> 3;   // 0..15
    const int i = threadIdx.x & 7;    // 0..7
    const int sb = r0 + r;

    float qr[HD];
    #pragma unroll
    for (int d = 0; d < HD; ++d) qr[d] = (float)sq[r][i][d];

    float sc[NB];
    #pragma unroll
    for (int j = 0; j < NB; ++j) {
        float s = 0.0f;
        #pragma unroll
        for (int d = 0; d < HD; ++d) s += qr[d] * (float)sk[r][j][d];
        sc[j] = s;
    }
    float mx = sc[0];
    #pragma unroll
    for (int j = 1; j < NB; ++j) mx = fmaxf(mx, sc[j]);
    float sum = 0.0f;
    #pragma unroll
    for (int j = 0; j < NB; ++j) { sc[j] = __expf(sc[j] - mx); sum += sc[j]; }
    const float inv = 1.0f / sum;
    #pragma unroll
    for (int j = 0; j < NB; ++j) sc[j] *= inv;

    // mean over heads -> atomic accumulate
    #pragma unroll
    for (int j = 0; j < NB; ++j)
        atomicAdd(&smean[((size_t)sb * NB + i) * NB + j], sc[j] * (1.0f / (float)NH));

    // out row = attn @ v, store f16 in (sb, block, h*64+d) layout
    _Float16* orow = attnb + ((size_t)sb * NB + i) * INNER + h * HD;
    #pragma unroll
    for (int c = 0; c < 8; ++c) {
        float o[8] = {0,0,0,0,0,0,0,0};
        #pragma unroll
        for (int j = 0; j < NB; ++j) {
            const float p = sc[j];
            #pragma unroll
            for (int dd = 0; dd < 8; ++dd)
                o[dd] += p * (float)sv[r][j][c * 8 + dd];
        }
        union { v8h v; _Float16 e[8]; } u;
        #pragma unroll
        for (int dd = 0; dd < 8; ++dd) u.e[dd] = (_Float16)o[dd];
        *(v8h*)(orow + c * 8) = u.v;
    }
}

// ---------------- Final grouped projection ----------------
// 2048 wave-tasks: (256 row tiles) x (8 blocks). block=256 (8 waves), grid=256.
__global__ __launch_bounds__(256) void proj_kernel(
    const _Float16* __restrict__ attnb,    // (4096, 8, 768)
    const _Float16* __restrict__ wfT,      // (8, 128, 768)
    float* __restrict__ out)               // (4096, 1024)
{
    const int wid  = blockIdx.x * 8 + (threadIdx.x >> 5);
    const int lane = threadIdx.x & 31;
    const int mt = wid >> 3;   // row tile
    const int kb = wid & 7;    // block
    const int r0 = mt * 16;

    const _Float16* abase = attnb + ((size_t)r0 * NB + kb) * INNER;
    const int lda = NB * INNER;                            // 6144 halves between sb rows
    const _Float16* bbase = wfT + (size_t)kb * BD * INNER; // 128 rows x 768 K

    v8f acc[8] = {};
    for (int ks = 0; ks < 24; ++ks) {
        v16h a = load_frag_a(abase + ks * 32, lda, lane);
        #pragma unroll
        for (int nt = 0; nt < 8; ++nt) {
            v16h b = load_frag_b(bbase + (size_t)(nt * 16) * INNER + ks * 32, INNER, lane);
            acc[nt] = __builtin_amdgcn_wmma_f32_16x16x32_f16(
                false, a, false, b, (short)0, acc[nt], false, false);
        }
    }
    const int col = lane & 15, hi = lane >> 4;
    #pragma unroll
    for (int nt = 0; nt < 8; ++nt) {
        #pragma unroll
        for (int g = 0; g < 8; ++g) {
            const int row = hi * 8 + g;
            out[(size_t)(r0 + row) * DIM + kb * BD + nt * 16 + col] = acc[nt][g];
        }
    }
}

// ---------------- Launch ----------------
extern "C" void kernel_launch(void* const* d_in, const int* in_sizes, int n_in,
                              void* d_out, int out_size, void* d_ws, size_t ws_size,
                              hipStream_t stream) {
    const float* x  = (const float*)d_in[0];
    const float* Wq = (const float*)d_in[1];
    const float* Wk = (const float*)d_in[2];
    const float* Wv = (const float*)d_in[3];
    const float* Wf = (const float*)d_in[4];

    float* out   = (float*)d_out;                 // (4096,1024)
    float* smean = out + (size_t)SB * DIM;        // (4096,8,8)

    // workspace layout (halves)
    _Float16* ws   = (_Float16*)d_ws;
    _Float16* xh   = ws;                                   // 4096*1024
    _Float16* wqT  = xh  + (size_t)SB * DIM;               // 8*768*128
    _Float16* wkT  = wqT + (size_t)NB * INNER * BD;
    _Float16* wvT  = wkT + (size_t)NB * INNER * BD;
    _Float16* wfT  = wvT + (size_t)NB * INNER * BD;
    _Float16* attnb= wfT + (size_t)NB * INNER * BD;        // 4096*8*768

    const int nSmean = SB * NB * NB;      // 262144
    const int nX     = SB * DIM;          // 4194304
    const int nW     = NB * BD * INNER;   // 786432

    zero_f32<<<(nSmean + 255) / 256, 256, 0, stream>>>(smean, nSmean);
    cvt_f32_f16<<<(nX + 255) / 256, 256, 0, stream>>>(x, xh, nX);
    transpose_w_qkv<<<(nW + 255) / 256, 256, 0, stream>>>(Wq, wqT);
    transpose_w_qkv<<<(nW + 255) / 256, 256, 0, stream>>>(Wk, wkT);
    transpose_w_qkv<<<(nW + 255) / 256, 256, 0, stream>>>(Wv, wvT);
    transpose_wf  <<<(nW + 255) / 256, 256, 0, stream>>>(Wf, wfT);

    dim3 g1(SB / 16, NH);   // 256 x 12
    qkv_attn_kernel<<<g1, 128, 0, stream>>>(xh, wqT, wkT, wvT, attnb, smean);

    proj_kernel<<<(SB / 16) * NB / 8, 256, 0, stream>>>(attnb, wfT, out);
}
